// RealTimeMIProcessor_111669150025
// MI455X (gfx1250) — compile-verified, matrix operand order checked by reading, not run on hardware
//
#include <hip/hip_runtime.h>

#define NROWS 12
#define NCOLS 12
#define NPATCH (NROWS * NCOLS)
#define PHh 32
#define PWw 32
#define PSIZE (PHh * PWw)
#define NSEL 36
#define NCAND 25
#define BINSK 64
#define IMW 384
#define HW (IMW * IMW)

typedef unsigned char u8;
typedef int v8i __attribute__((ext_vector_type(8)));

// ---------------- Kernel 1: per-batch min / max ----------------
__global__ __launch_bounds__(256) void mi_minmax(const float* __restrict__ sino,
                                                 float* __restrict__ mn,
                                                 float* __restrict__ mx) {
  __shared__ float smin[256], smax[256];
  const int b = blockIdx.x, tid = threadIdx.x;
  const float* p = sino + (size_t)b * HW;
  float lo = 3.402823466e+38f, hi = -3.402823466e+38f;
  for (int i = tid; i < HW; i += 256) {
    float v = p[i];
    lo = fminf(lo, v);
    hi = fmaxf(hi, v);
  }
  smin[tid] = lo;
  smax[tid] = hi;
  __syncthreads();
  for (int s = 128; s > 0; s >>= 1) {
    if (tid < s) {
      smin[tid] = fminf(smin[tid], smin[tid + s]);
      smax[tid] = fmaxf(smax[tid], smax[tid + s]);
    }
    __syncthreads();
  }
  if (tid == 0) { mn[b] = smin[0]; mx[b] = smax[0]; }
}

// ------- Kernel 2: quantize to 64 bins, repack patch-major u8 -------
__global__ __launch_bounds__(256) void mi_bin(const float* __restrict__ sino,
                                              const float* __restrict__ mn,
                                              const float* __restrict__ mx,
                                              u8* __restrict__ binned) {
  const int p = blockIdx.x, b = blockIdx.y, tid = threadIdx.x;
  const float lo = mn[b];
  const float inv = 1.0f / (mx[b] - lo + 1e-6f);
  const int prow = p / NCOLS, pcol = p % NCOLS;
  const float* src = sino + (size_t)b * HW + (size_t)(prow * PHh) * IMW + pcol * PWw;
  u8* dst = binned + ((size_t)b * NPATCH + p) * PSIZE;
  for (int i = tid; i < PSIZE; i += 256) {
    const int ph = i >> 5, pw = i & 31;
    float x = src[ph * IMW + pw];
    float nrm = (x - lo) * inv;
    nrm = fminf(fmaxf(nrm, 0.0f), 1.0f);
    int bin = (int)(nrm * 63.0f);  // truncation, matches astype(int32)
    bin = min(max(bin, 0), 63);
    dst[i] = (u8)bin;
  }
}

// ------- Kernel 3: distances (IU8 WMMA), argmin, joint hist, MI -------
__global__ __launch_bounds__(256) void mi_main(const u8* __restrict__ binned,
                                               float* __restrict__ out) {
  __shared__ u8 s_targ[PSIZE];
  __shared__ u8 s_cand[NCAND * PSIZE];
  __shared__ int s_cidx[NCAND];
  __shared__ int s_part[8][16];
  __shared__ int s_dist[NCAND];
  __shared__ int s_best;
  __shared__ unsigned s_hist[BINSK * BINSK];
  __shared__ float s_red[256];
  __shared__ float s_marg[2 * BINSK];
  __shared__ float s_h[3];

  const int sel = blockIdx.x, b = blockIdx.y, tid = threadIdx.x;
  const int t_row = (sel / 6) * 2, t_col = (sel % 6) * 2;
  const int tp = t_row * NCOLS + t_col;

  // candidate window indices (clipped) — order matches reference (dr outer, dc inner)
  if (tid < NCAND) {
    const int dr = tid / 5 - 2, dc = tid % 5 - 2;
    const int rr = min(max(t_row + dr, 0), NROWS - 1);
    const int cc = min(max(t_col + dc, 0), NCOLS - 1);
    s_cidx[tid] = rr * NCOLS + cc;
  }
  __syncthreads();

  // ---- stage target + 25 candidate patches straight into LDS via async copy ----
  const u8* base = binned + (size_t)b * NPATCH * PSIZE;
  {
    const int TSEG = PSIZE / 16;           // 64 x 16B segments for the target
    const int NSEG = TSEG + NCAND * TSEG;  // 1664 segments total
    for (int i = tid; i < NSEG; i += 256) {
      int pidx, off;
      u8* dst;
      if (i < TSEG) {
        pidx = tp;
        off = i;
        dst = s_targ + (size_t)i * 16;
      } else {
        const int j = i - TSEG;
        pidx = s_cidx[j >> 6];
        off = j & 63;
        dst = s_cand + (size_t)j * 16;
      }
      const unsigned ldsb = (unsigned)(unsigned long long)dst;  // LDS byte address
      const unsigned long long ga =
          (unsigned long long)(base + (size_t)pidx * PSIZE + (size_t)off * 16);
      // CDNA5 direct global->LDS async copy (ASYNCcnt), no VGPR bounce
      asm volatile("global_load_async_to_lds_b128 %0, %1, off"
                   :: "v"(ldsb), "v"(ga)
                   : "memory");
    }
    asm volatile("s_wait_asynccnt 0x0" ::: "memory");
  }
  __syncthreads();

  // ---- L1 distances: D = |cand - targ| x ones  via V_WMMA_I32_16X16X64_IU8 ----
  const int wave = tid >> 5, lane = tid & 31;
  const int mrow = lane & 15, hiK = lane >> 4;
  const int tile = wave >> 2;  // waves 0-3: cands 0-15, waves 4-7: 16-24
  const int cand = tile * 16 + mrow;
  // pad rows (>=25) alias the target so their diffs are zero (branchless, EXEC stays full)
  const u8* cp = (cand < NCAND) ? (s_cand + (size_t)cand * PSIZE) : s_targ;

  v8i onesB = {0x01010101, 0x01010101, 0x01010101, 0x01010101,
               0x01010101, 0x01010101, 0x01010101, 0x01010101};
  v8i acc = {0, 0, 0, 0, 0, 0, 0, 0};

  for (int chunk = (wave & 3); chunk < 16; chunk += 4) {
    v8i a;
    // ISA 8-bit A 16x64 layout: lane<16 holds K = j*16+[0..7] in VGPR pair 2j; lanes 16-31: +8
#pragma unroll
    for (int j = 0; j < 4; ++j) {
      const int k0 = chunk * 64 + j * 16 + hiK * 8;
      const uint2 cv = *(const uint2*)(cp + k0);
      const uint2 tv = *(const uint2*)(s_targ + k0);
      unsigned r0 = 0, r1 = 0;
#pragma unroll
      for (int t = 0; t < 4; ++t) {
        int d0 = (int)((cv.x >> (8 * t)) & 0xFF) - (int)((tv.x >> (8 * t)) & 0xFF);
        d0 = d0 < 0 ? -d0 : d0;
        r0 |= (unsigned)d0 << (8 * t);
        int d1 = (int)((cv.y >> (8 * t)) & 0xFF) - (int)((tv.y >> (8 * t)) & 0xFF);
        d1 = d1 < 0 ? -d1 : d1;
        r1 |= (unsigned)d1 << (8 * t);
      }
      a[2 * j] = (int)r0;
      a[2 * j + 1] = (int)r1;
    }
    // (sgn_a, A, sgn_b, B, C, reuse_a, reuse_b)
    acc = __builtin_amdgcn_wmma_i32_16x16x64_iu8(false, a, false, onesB, acc, false, false);
  }

  // D layout: VGPR r, lanes 0-15 -> M=r ; lanes 16-31 -> M=8+r (all N columns identical)
  if (lane == 0 || lane == 16) {
    const int rbase = hiK * 8;
#pragma unroll
    for (int r = 0; r < 8; ++r) s_part[wave][rbase + r] = acc[r];
  }
  __syncthreads();

  if (tid < NCAND) {
    int d;
    if (tid < 16) {
      d = s_part[0][tid] + s_part[1][tid] + s_part[2][tid] + s_part[3][tid];
    } else {
      const int m = tid - 16;
      d = s_part[4][m] + s_part[5][m] + s_part[6][m] + s_part[7][m];
    }
    if (s_cidx[tid] == tp) d = 0x7FFFFFFF;  // self mask (ref adds 1e9 >> max dist 64512)
    s_dist[tid] = d;
  }
  __syncthreads();
  if (tid == 0) {
    int bd = s_dist[0], bi = 0;
    for (int c = 1; c < NCAND; ++c)
      if (s_dist[c] < bd) { bd = s_dist[c]; bi = c; }  // first-min, matches jnp.argmin
    s_best = bi;
  }
  for (int i = tid; i < BINSK * BINSK; i += 256) s_hist[i] = 0u;
  __syncthreads();

  // ---- joint histogram in LDS ----
  {
    const u8* bp = s_cand + (size_t)s_best * PSIZE;
    for (int i = tid; i < PSIZE; i += 256)
      atomicAdd(&s_hist[(int)s_targ[i] * BINSK + (int)bp[i]], 1u);
  }
  __syncthreads();

  // ---- entropies ----
  const float invP = 1.0f / (float)PSIZE;
  float hxy = 0.0f;
  for (int i = tid; i < BINSK * BINSK; i += 256) {
    const float p = (float)s_hist[i] * invP;
    hxy += p * logf(p + 1e-8f);
  }
  s_red[tid] = hxy;
  __syncthreads();
  for (int s = 128; s > 0; s >>= 1) {
    if (tid < s) s_red[tid] += s_red[tid + s];
    __syncthreads();
  }
  if (tid == 0) s_h[2] = -s_red[0];  // h_xy
  __syncthreads();

  if (tid < 2 * BINSK) {
    const int r = tid & 63, isY = tid >> 6;
    unsigned s = 0;
    for (int c = 0; c < BINSK; ++c)
      s += isY ? s_hist[c * BINSK + r] : s_hist[r * BINSK + c];
    s_marg[tid] = (float)s * invP;  // exact: k/1024 in f32
  }
  __syncthreads();
  float hterm = 0.0f;
  if (tid < 2 * BINSK) {
    const float p = s_marg[tid];
    hterm = p * logf(p + 1e-8f);
  }
  s_red[tid] = (tid < BINSK) ? hterm : 0.0f;
  __syncthreads();
  for (int s = 128; s > 0; s >>= 1) {
    if (tid < s) s_red[tid] += s_red[tid + s];
    __syncthreads();
  }
  if (tid == 0) s_h[0] = -s_red[0];  // h_x
  __syncthreads();
  s_red[tid] = (tid >= BINSK && tid < 2 * BINSK) ? hterm : 0.0f;
  __syncthreads();
  for (int s = 128; s > 0; s >>= 1) {
    if (tid < s) s_red[tid] += s_red[tid + s];
    __syncthreads();
  }
  if (tid == 0) {
    const float h_y = -s_red[0];
    float mi = s_h[0] + h_y - s_h[2];
    if (mi != mi) mi = 0.0f;  // NaN -> 0
    out[b * NSEL + sel] = log1pf(mi);
  }
}

extern "C" void kernel_launch(void* const* d_in, const int* in_sizes, int n_in,
                              void* d_out, int out_size, void* d_ws, size_t ws_size,
                              hipStream_t stream) {
  (void)n_in; (void)out_size; (void)ws_size;
  const float* sino = (const float*)d_in[0];
  float* out = (float*)d_out;
  const int B = in_sizes[0] / HW;  // 64

  float* mn = (float*)d_ws;
  float* mx = mn + B;
  u8* binned = (u8*)d_ws + 512;  // 16B-aligned; B*144*1024 = 9.44 MB

  mi_minmax<<<dim3(B), dim3(256), 0, stream>>>(sino, mn, mx);
  mi_bin<<<dim3(NPATCH, B), dim3(256), 0, stream>>>(sino, mn, mx, binned);
  mi_main<<<dim3(NSEL, B), dim3(256), 0, stream>>>(binned, out);
}
